// MultiHeadAttention_77455440216748
// MI455X (gfx1250) — matrix operand order repaired and checked by reading, not compiled, not run on hardware
//
#include <hip/hip_runtime.h>
#include <hip/hip_bf16.h>

// MHA forward for gfx1250 (MI455X), bf16 WMMA with f32 accumulation.
// Workspace layout (needs ~88 MB of d_ws):
//   xb[B*S*D] bf16, Wt{q,k,v}[H,HD,D] bf16 (transposed), Wo_b[D,D] bf16,
//   Q[B,H,S,HD] bf16 (pre-scaled by 1/sqrt(HD)), K[B,H,S,HD] bf16,
//   Vt[B,H,HD,S] bf16, attn[B,S,D] bf16.

#define BB 4
#define SS 2048
#define DD 1024
#define HH 16
#define HD 64

typedef __attribute__((ext_vector_type(16))) __bf16 v16bf;
typedef __attribute__((ext_vector_type(8)))  float  v8f;
typedef __attribute__((ext_vector_type(4)))  unsigned int u32x4;

union AF { v16bf v; u32x4 q[2]; unsigned short u[16]; };

#define WMMA_BF16(a, b, c) \
    __builtin_amdgcn_wmma_f32_16x16x32_bf16(false, (a).v, false, (b).v, (short)0, (c), false, false)

__device__ __forceinline__ unsigned short f2bf(float f) {
    unsigned int x = __float_as_uint(f);
    x += 0x7FFFu + ((x >> 16) & 1u);         // round-to-nearest-even
    return (unsigned short)(x >> 16);
}

__device__ __forceinline__ u32x4 ldg4(const unsigned short* p) {
    return *reinterpret_cast<const u32x4*>(p);   // global_load_b128
}

// ---------------- fp32 -> bf16 elementwise ----------------
__global__ void cvt_bf16(const float* __restrict__ in,
                         unsigned short* __restrict__ out, int n) {
    int i = blockIdx.x * blockDim.x + threadIdx.x;
    if (i < n) out[i] = f2bf(in[i]);
}

// ---------------- Wq/Wk/Wv [H,D,HD] fp32 -> [H,HD,D] bf16 (transpose) ----------------
__global__ void cvt_wqkv(const float* __restrict__ in,
                         unsigned short* __restrict__ out) {
    int i = blockIdx.x * blockDim.x + threadIdx.x;   // over H*HD*D, out-major
    int d = i % DD;
    int e = (i / DD) % HD;
    int h = i / (DD * HD);
    out[i] = f2bf(in[((size_t)h * DD + d) * HD + e]);
}

// ---------------- QKV projection: 32(s) x 64(e) tile per wave ----------------
// transposed==0: out = [B,H,S,HD] bf16 ; transposed==1: out = [B,H,HD,S] bf16 (for V)
// oscale folded into the bf16 store (1/sqrt(HD) for Q, 1.0 for K/V).
__global__ void proj_qkv(const unsigned short* __restrict__ xb,
                         const unsigned short* __restrict__ wt,
                         unsigned short* __restrict__ out,
                         int transposed, float oscale) {
    int w    = threadIdx.x >> 5;
    int lane = threadIdx.x & 31;
    int n  = lane & 15, hi = lane >> 4;
    int bh = blockIdx.y;
    int b  = bh / HH, h = bh % HH;
    int i0 = blockIdx.x * 128 + w * 32;

    const unsigned short* xrow0 = xb + ((size_t)b * SS + i0 + n) * DD;
    const unsigned short* xrow1 = xrow0 + (size_t)16 * DD;
    v8f acc[2][4] = {};

    for (int k0 = 0; k0 < DD; k0 += 32) {
        AF a0, a1;
        a0.q[0] = ldg4(xrow0 + k0 + 8 * hi);
        a0.q[1] = ldg4(xrow0 + k0 + 16 + 8 * hi);
        a1.q[0] = ldg4(xrow1 + k0 + 8 * hi);
        a1.q[1] = ldg4(xrow1 + k0 + 16 + 8 * hi);
#pragma unroll
        for (int t = 0; t < 4; ++t) {
            const unsigned short* wrow =
                wt + ((size_t)h * HD + t * 16 + n) * DD + k0 + 8 * hi;
            AF bw;
            bw.q[0] = ldg4(wrow);
            bw.q[1] = ldg4(wrow + 16);
            acc[0][t] = WMMA_BF16(a0, bw, acc[0][t]);   // B-frag reused for both row tiles
            acc[1][t] = WMMA_BF16(a1, bw, acc[1][t]);
        }
    }
#pragma unroll
    for (int r = 0; r < 2; ++r)
#pragma unroll
        for (int t = 0; t < 4; ++t)
#pragma unroll
            for (int j = 0; j < 8; ++j) {
                int row = i0 + r * 16 + 8 * hi + j;
                int e   = t * 16 + n;
                unsigned short val = f2bf(acc[r][t][j] * oscale);
                if (transposed)
                    out[(size_t)bh * HD * SS + (size_t)e * SS + row] = val;
                else
                    out[((size_t)bh * SS + row) * HD + e] = val;
            }
}

// ---------------- flash attention: per-wave 16 query rows, 32-key steps ----------------
__global__ void flash_attn(const unsigned short* __restrict__ qb,
                           const unsigned short* __restrict__ kb,
                           const unsigned short* __restrict__ vt,
                           unsigned short* __restrict__ attnb) {
    __shared__ __align__(16) unsigned short plds[4][16 * 40];  // 80-B padded rows

    int w    = threadIdx.x >> 5;
    int lane = threadIdx.x & 31;
    int n  = lane & 15, hi = lane >> 4;
    int bh = blockIdx.y;
    int b  = bh / HH, h = bh % HH;
    int i0 = blockIdx.x * 64 + w * 16;

    const unsigned short* qbase = qb + (size_t)bh * SS * HD;
    const unsigned short* kbase = kb + (size_t)bh * SS * HD;
    const unsigned short* vbase = vt + (size_t)bh * HD * SS;

    AF aq[2];
    {
        const unsigned short* qrow = qbase + (size_t)(i0 + n) * HD;
        aq[0].q[0] = ldg4(qrow + 8 * hi);
        aq[0].q[1] = ldg4(qrow + 16 + 8 * hi);
        aq[1].q[0] = ldg4(qrow + 32 + 8 * hi);
        aq[1].q[1] = ldg4(qrow + 48 + 8 * hi);
    }

    float mrow[8], lrow[8];
    v8f acc[4] = {};
#pragma unroll
    for (int j = 0; j < 8; ++j) { mrow[j] = -__builtin_inff(); lrow[j] = 0.f; }

    const int ihi = i0 + 15;
    for (int j0 = 0; j0 <= ihi; j0 += 32) {
        // --- scores: two 16x16 tiles over keys [j0,j0+16) and [j0+16,j0+32)
        v8f sc[2];
#pragma unroll
        for (int sub = 0; sub < 2; ++sub) {
            const unsigned short* krow = kbase + (size_t)(j0 + sub * 16 + n) * HD;
            AF b0, b1;
            b0.q[0] = ldg4(krow + 8 * hi);
            b0.q[1] = ldg4(krow + 16 + 8 * hi);
            b1.q[0] = ldg4(krow + 32 + 8 * hi);
            b1.q[1] = ldg4(krow + 48 + 8 * hi);
            v8f s = {};
            s = WMMA_BF16(aq[0], b0, s);
            s = WMMA_BF16(aq[1], b1, s);
            sc[sub] = s;
        }
        // --- causal mask (only needed near the diagonal; wave-uniform test)
        const bool needmask = (j0 + 31 > i0);
        // --- online softmax (reductions over the 16-lane halves; wave32-correct)
#pragma unroll
        for (int j = 0; j < 8; ++j) {
            float slo = sc[0][j];    // Q pre-scaled by 1/sqrt(HD)
            float shi = sc[1][j];
            if (needmask) {
                int qi = i0 + 8 * hi + j;
                if (j0 + n > qi)      slo = -__builtin_inff();
                if (j0 + 16 + n > qi) shi = -__builtin_inff();
            }
            float tm = fmaxf(slo, shi);
            tm = fmaxf(tm, __shfl_xor(tm, 1));
            tm = fmaxf(tm, __shfl_xor(tm, 2));
            tm = fmaxf(tm, __shfl_xor(tm, 4));
            tm = fmaxf(tm, __shfl_xor(tm, 8));
            float newm  = fmaxf(mrow[j], tm);
            float alpha = __expf(mrow[j] - newm);
            float plo = __expf(slo - newm);
            float phi = __expf(shi - newm);
            float rs = plo + phi;
            rs += __shfl_xor(rs, 1);
            rs += __shfl_xor(rs, 2);
            rs += __shfl_xor(rs, 4);
            rs += __shfl_xor(rs, 8);
            lrow[j] = lrow[j] * alpha + rs;
            mrow[j] = newm;
#pragma unroll
            for (int t = 0; t < 4; ++t) acc[t][j] *= alpha;
            int r = (8 * hi + j) * 40;
            plds[w][r + n]      = f2bf(plo);
            plds[w][r + 16 + n] = f2bf(phi);
        }
        // --- P: C-layout -> A-frag via LDS (same-wave DS ordering, no barrier)
        AF ap;
        ap.q[0] = *reinterpret_cast<const u32x4*>(&plds[w][n * 40 + 8 * hi]);
        ap.q[1] = *reinterpret_cast<const u32x4*>(&plds[w][n * 40 + 16 + 8 * hi]);
        // --- PV: acc += P(16x32) x V(32x64), V stored transposed [HD,S]
#pragma unroll
        for (int t = 0; t < 4; ++t) {
            const unsigned short* vrow =
                vbase + (size_t)(t * 16 + n) * SS + j0 + 8 * hi;
            AF bv;
            bv.q[0] = ldg4(vrow);
            bv.q[1] = ldg4(vrow + 16);
            acc[t] = WMMA_BF16(ap, bv, acc[t]);
        }
    }
    // --- finalize (divide by row sum) and emit head-concatenated bf16
#pragma unroll
    for (int j = 0; j < 8; ++j) {
        float inv = 1.0f / lrow[j];
        int row = i0 + 8 * hi + j;
#pragma unroll
        for (int t = 0; t < 4; ++t)
            attnb[((size_t)b * SS + row) * DD + h * HD + t * 16 + n] =
                f2bf(acc[t][j] * inv);
    }
}

// ---------------- output projection: 32(s) x 64(o) tile per wave ----------------
__global__ void out_proj(const unsigned short* __restrict__ attnb,
                         const unsigned short* __restrict__ wob,
                         const float* __restrict__ bo,
                         float* __restrict__ out) {
    int w    = threadIdx.x >> 5;
    int lane = threadIdx.x & 31;
    int n  = lane & 15, hi = lane >> 4;
    int o0 = blockIdx.x * 64;
    int r0 = blockIdx.y * 128 + w * 32;   // flat row b*S+s

    const unsigned short* arow0 = attnb + (size_t)(r0 + n) * DD;
    const unsigned short* arow1 = arow0 + (size_t)16 * DD;
    v8f acc[2][4] = {};

    for (int k0 = 0; k0 < DD; k0 += 32) {
        AF a0, a1;
        a0.q[0] = ldg4(arow0 + k0 + 8 * hi);
        a0.q[1] = ldg4(arow0 + k0 + 16 + 8 * hi);
        a1.q[0] = ldg4(arow1 + k0 + 8 * hi);
        a1.q[1] = ldg4(arow1 + k0 + 16 + 8 * hi);
#pragma unroll
        for (int t = 0; t < 4; ++t) {
            const unsigned short* wrow =
                wob + (size_t)(o0 + t * 16 + n) * DD + k0 + 8 * hi;  // torch [out,in]
            AF bw;
            bw.q[0] = ldg4(wrow);
            bw.q[1] = ldg4(wrow + 16);
            acc[0][t] = WMMA_BF16(a0, bw, acc[0][t]);
            acc[1][t] = WMMA_BF16(a1, bw, acc[1][t]);
        }
    }
#pragma unroll
    for (int t = 0; t < 4; ++t) {
        float bias = bo[o0 + t * 16 + n];
#pragma unroll
        for (int r = 0; r < 2; ++r)
#pragma unroll
            for (int j = 0; j < 8; ++j)
                out[(size_t)(r0 + r * 16 + 8 * hi + j) * DD + o0 + t * 16 + n] =
                    acc[r][t][j] + bias;
    }
}

extern "C" void kernel_launch(void* const* d_in, const int* in_sizes, int n_in,
                              void* d_out, int out_size, void* d_ws, size_t ws_size,
                              hipStream_t stream) {
    (void)in_sizes; (void)n_in; (void)out_size; (void)ws_size;
    const float* x  = (const float*)d_in[0];
    const float* Wq = (const float*)d_in[1];
    const float* Wk = (const float*)d_in[2];
    const float* Wv = (const float*)d_in[3];
    const float* Wo = (const float*)d_in[4];
    const float* bo = (const float*)d_in[5];
    float* out = (float*)d_out;

    char* ws = (char*)d_ws;
    const size_t XB  = (size_t)BB * SS * DD * 2;        // 16 MB
    const size_t WT  = (size_t)HH * DD * HD * 2;        //  2 MB
    const size_t QKV = (size_t)BB * HH * SS * HD * 2;   // 16 MB
    unsigned short* xb    = (unsigned short*)(ws);
    unsigned short* wtq   = (unsigned short*)(ws + XB);
    unsigned short* wtk   = (unsigned short*)(ws + XB + WT);
    unsigned short* wtv   = (unsigned short*)(ws + XB + 2 * WT);
    unsigned short* wob   = (unsigned short*)(ws + XB + 3 * WT);
    unsigned short* qbv   = (unsigned short*)(ws + XB + 4 * WT);
    unsigned short* kbv   = (unsigned short*)(ws + XB + 4 * WT + QKV);
    unsigned short* vtb   = (unsigned short*)(ws + XB + 4 * WT + 2 * QKV);
    unsigned short* attnb = (unsigned short*)(ws + XB + 4 * WT + 3 * QKV);

    int nx = BB * SS * DD;
    cvt_bf16<<<(nx + 255) / 256, 256, 0, stream>>>(x, xb, nx);
    cvt_bf16<<<(DD * DD + 255) / 256, 256, 0, stream>>>(Wo, wob, DD * DD);
    int nw = HH * HD * DD;
    cvt_wqkv<<<(nw + 255) / 256, 256, 0, stream>>>(Wq, wtq);
    cvt_wqkv<<<(nw + 255) / 256, 256, 0, stream>>>(Wk, wtk);
    cvt_wqkv<<<(nw + 255) / 256, 256, 0, stream>>>(Wv, wtv);

    dim3 pg(SS / 128, BB * HH);     // (16, 64), 4 waves/block, 32 rows/wave
    proj_qkv<<<pg, 128, 0, stream>>>(xb, wtq, qbv, 0, 0.125f);  // Q pre-scaled
    proj_qkv<<<pg, 128, 0, stream>>>(xb, wtk, kbv, 0, 1.0f);
    proj_qkv<<<pg, 128, 0, stream>>>(xb, wtv, vtb, 1, 1.0f);

    dim3 fg(SS / 64, BB * HH);      // (32, 64), 4 waves/block, 16 rows/wave
    flash_attn<<<fg, 128, 0, stream>>>(qbv, kbv, vtb, attnb);

    dim3 og(DD / 64, BB * SS / 128); // (16, 64)
    out_proj<<<og, 128, 0, stream>>>(attnb, wob, bo, out);
}
